// _PointnetSAModuleFSBase_78262894068237
// MI455X (gfx1250) — matrix operand order repaired and checked by reading, not compiled, not run on hardware
//
#include <hip/hip_runtime.h>
#include <hip/hip_bf16.h>

// ---------------------------------------------------------------------------
// PointNet++ SA module for MI455X (gfx1250, wave32, WMMA 16x16x32 f16)
// Pipeline: transpose feats -> pack weights f16 (N-major) -> FPS -> ball
//           query -> grouped MLP (WMMA) + maxpool -> agg GEMM (WMMA) + scores
// LDS WMMA fragments load via 2x ds_load_b128 per lane; agg A loads from global f32.
// ---------------------------------------------------------------------------

typedef __attribute__((ext_vector_type(16))) _Float16 v16h;
typedef __attribute__((ext_vector_type(8)))  _Float16 v8h;
typedef __attribute__((ext_vector_type(8)))  float    v8f;

#define BQ       2
#define NQ       16384
#define CIN      64
#define SQ       4096
#define NSAMP    32
#define RAD2     0.25f

// workspace layout (bytes)
#define WS_FEATST   0u                       // B*N*64 f32   = 8388608
#define WS_SIDX     8388608u                 // B*S   i32    = 32768
#define WS_BIDX     8421376u                 // B*S*32 i32   = 1048576
#define WS_POOLED   9469952u                 // B*S*128 f32  = 4194304
#define WS_W1P      13664256u                // 64x96  f16   = 12288   [n][k]
#define WS_W2P      13676544u                // 64x64  f16   = 8192    [n][k]
#define WS_W3P      13684736u                // 128x64 f16   = 16384   [n][k]
#define WS_WAP      13701120u                // 128x128 f16  = 32768   [n][k]
// total ~13.1 MB

// output layout (floats)
#define OUT_NEWXYZ  0
#define OUT_NEWFEAT (BQ*SQ*3)                           // 24576
#define OUT_SCORES  (BQ*SQ*3 + BQ*128*SQ)               // 1073152

__device__ __forceinline__ v8f zero8() {
  v8f z;
#pragma unroll
  for (int i = 0; i < 8; ++i) z[i] = 0.f;
  return z;
}

__device__ __forceinline__ v16h cat8(v8h lo, v8h hi) {
  return __builtin_shufflevector(lo, hi, 0, 1, 2, 3, 4, 5, 6, 7, 8, 9, 10, 11,
                                 12, 13, 14, 15);
}

__device__ __forceinline__ void lds_fence() {
  // per-wave LDS produce->consume ordering; also a compiler barrier
  asm volatile("s_wait_dscnt 0" ::: "memory");
}

// A fragment: 16x32 f16 tile from row-major [rows][ldk] LDS.
// lane<16: M=lane, K in {0..7,16..23}; lane>=16: M=lane-16, K in {8..15,24..31}
// -> per lane two contiguous runs of 8 halfs: 2x ds_load_b128
__device__ __forceinline__ v16h load_a_frag(const _Float16* A, int row_base,
                                            int k_base, int ldk) {
  const int l  = threadIdx.x & 31;
  const int hh = l >> 4;
  const int r  = (l & 15) + row_base;
  const _Float16* p = A + r * ldk + k_base + hh * 8;
  v8h lo = *(const v8h*)p;
  v8h hi = *(const v8h*)(p + 16);
  return cat8(lo, hi);
}

// B fragment: 32x16 f16 tile from N-major [n][ldk] LDS.
// lane<16: N=lane, K in {0..7,16..23}; lane>=16: N=lane-16, K in {8..15,24..31}
// -> per lane two contiguous runs of 8 halfs: 2x ds_load_b128
__device__ __forceinline__ v16h load_b_frag(const _Float16* Bm, int k_base,
                                            int n_base, int ldk) {
  const int l  = threadIdx.x & 31;
  const int hh = l >> 4;
  const int n  = (l & 15) + n_base;
  const _Float16* p = Bm + n * ldk + k_base + hh * 8;
  v8h lo = *(const v8h*)p;
  v8h hi = *(const v8h*)(p + 16);
  return cat8(lo, hi);
}

// C/D 16x16 f32: VGPR v -> M = v + 8*(lane/16), N = lane%16
__device__ __forceinline__ void store_c_relu_f16(_Float16* H, v8f acc,
                                                 const float* bias,
                                                 int row_base, int n_base,
                                                 int ldn) {
  const int l  = threadIdx.x & 31;
  const int hh = l >> 4;
  const int n  = (l & 15) + n_base;
  const float bv = bias[n];
#pragma unroll
  for (int v = 0; v < 8; ++v) {
    float x = acc[v] + bv;
    x = x > 0.f ? x : 0.f;
    H[(row_base + v + hh * 8) * ldn + n] = (_Float16)x;
  }
}

// ---------------------------------------------------------------------------
// K0: features (B,C,N) -> featsT (B,N,C), coalesced writes
// ---------------------------------------------------------------------------
__global__ void k_transpose_feats(const float* __restrict__ feats,
                                  float* __restrict__ featsT) {
  int gid = blockIdx.x * blockDim.x + threadIdx.x;
  if (gid >= BQ * NQ * CIN) return;
  int c = gid & (CIN - 1);
  int n = (gid / CIN) & (NQ - 1);
  int b = gid / (CIN * NQ);
  featsT[gid] = feats[b * CIN * NQ + c * NQ + n];
}

// ---------------------------------------------------------------------------
// K1: pack weights to f16 N-major [n][K].
//     Layer-1 K order is permuted: K 0..63 = features, K 64..66 = xyz, pad..95
//     (matches the A-tile gather layout; w2/w3/wa are identity converts).
// ---------------------------------------------------------------------------
__global__ void k_pack_weights(const float* __restrict__ w1,
                               const float* __restrict__ w2,
                               const float* __restrict__ w3,
                               const float* __restrict__ wa,
                               _Float16* __restrict__ W1p,
                               _Float16* __restrict__ W2p,
                               _Float16* __restrict__ W3p,
                               _Float16* __restrict__ WAp) {
  const int total = 64 * 96 + 64 * 64 + 128 * 64 + 128 * 128;
  for (int i = blockIdx.x * blockDim.x + threadIdx.x; i < total;
       i += gridDim.x * blockDim.x) {
    if (i < 6144) {
      int n = i / 96, k = i % 96;
      float v = 0.f;
      if (k < 64)      v = w1[n * 67 + 3 + k];   // feature weights
      else if (k < 67) v = w1[n * 67 + (k - 64)]; // xyz weights
      W1p[i] = (_Float16)v;
    } else if (i < 10240) {
      W2p[i - 6144] = (_Float16)w2[i - 6144];     // [n][k] == row-major (o,i)
    } else if (i < 18432) {
      W3p[i - 10240] = (_Float16)w3[i - 10240];
    } else {
      WAp[i - 18432] = (_Float16)wa[i - 18432];
    }
  }
}

// ---------------------------------------------------------------------------
// K2: farthest point sampling. One 1024-thread block per batch.
//     Points live in registers (16/thread); argmax via shuffle + LDS tree.
// ---------------------------------------------------------------------------
__global__ void __launch_bounds__(1024)
k_fps(const float* __restrict__ xyz, int* __restrict__ sidx,
      float* __restrict__ new_xyz) {
  const int b = blockIdx.x;
  const int t = threadIdx.x;
  const int lane = t & 31, wid = t >> 5;

  __shared__ float s_val[32];
  __shared__ int   s_idx[32];
  __shared__ float s_cent[3];
  __shared__ int   s_far;

  float px[16], py[16], pz[16], dist[16];
#pragma unroll
  for (int j = 0; j < 16; ++j) {
    int i = j * 1024 + t;
    px[j] = xyz[(b * NQ + i) * 3 + 0];
    py[j] = xyz[(b * NQ + i) * 3 + 1];
    pz[j] = xyz[(b * NQ + i) * 3 + 2];
    dist[j] = 1e10f;
  }

  int far = 0;
  for (int s = 0; s < SQ; ++s) {
    if (t == 0) {
      sidx[b * SQ + s] = far;
      float cx = xyz[(b * NQ + far) * 3 + 0];
      float cy = xyz[(b * NQ + far) * 3 + 1];
      float cz = xyz[(b * NQ + far) * 3 + 2];
      s_cent[0] = cx; s_cent[1] = cy; s_cent[2] = cz;
      new_xyz[(b * SQ + s) * 3 + 0] = cx;
      new_xyz[(b * SQ + s) * 3 + 1] = cy;
      new_xyz[(b * SQ + s) * 3 + 2] = cz;
    }
    __syncthreads();
    const float cx = s_cent[0], cy = s_cent[1], cz = s_cent[2];

    float bestv = -1.f; int besti = 0;
#pragma unroll
    for (int j = 0; j < 16; ++j) {
      float dx = px[j] - cx, dy = py[j] - cy, dz = pz[j] - cz;
      float d = dx * dx + dy * dy + dz * dz;
      dist[j] = fminf(dist[j], d);
      if (dist[j] > bestv) { bestv = dist[j]; besti = j * 1024 + t; }
    }
#pragma unroll
    for (int off = 16; off >= 1; off >>= 1) {
      float ov = __shfl_xor(bestv, off, 32);
      int   oi = __shfl_xor(besti, off, 32);
      if (ov > bestv || (ov == bestv && oi < besti)) { bestv = ov; besti = oi; }
    }
    if (lane == 0) { s_val[wid] = bestv; s_idx[wid] = besti; }
    __syncthreads();
    if (wid == 0) {
      float v = s_val[lane]; int i2 = s_idx[lane];
#pragma unroll
      for (int off = 16; off >= 1; off >>= 1) {
        float ov = __shfl_xor(v, off, 32);
        int   oi = __shfl_xor(i2, off, 32);
        if (ov > v || (ov == v && oi < i2)) { v = ov; i2 = oi; }
      }
      if (lane == 0) s_far = i2;
    }
    __syncthreads();
    far = s_far;
  }
}

// ---------------------------------------------------------------------------
// K3: ball query. One wave per sample point; ordered first-32-in-radius
//     collection via ballot + popcount (matches top_k(-keys) semantics).
// ---------------------------------------------------------------------------
__global__ void k_ball_query(const float* __restrict__ xyz,
                             const float* __restrict__ new_xyz,
                             int* __restrict__ bidx) {
  const int lane = threadIdx.x & 31;
  const int wid  = threadIdx.x >> 5;
  const int sample = blockIdx.x * 8 + wid;
  const int b = sample / SQ, s = sample % SQ;

  __shared__ int slots[8][NSAMP];

  const float qx = new_xyz[(b * SQ + s) * 3 + 0];
  const float qy = new_xyz[(b * SQ + s) * 3 + 1];
  const float qz = new_xyz[(b * SQ + s) * 3 + 2];

  int count = 0, first = -1;
  for (int base = 0; base < NQ; base += 32) {
    int i = base + lane;
    float dx = xyz[(b * NQ + i) * 3 + 0] - qx;
    float dy = xyz[(b * NQ + i) * 3 + 1] - qy;
    float dz = xyz[(b * NQ + i) * 3 + 2] - qz;
    bool inb = (dx * dx + dy * dy + dz * dz) < RAD2;
    unsigned mask = (unsigned)__ballot(inb);
    if (inb) {
      int pos = count + __popc(mask & ((1u << lane) - 1u));
      if (pos < NSAMP) slots[wid][pos] = i;
    }
    if (first < 0 && mask) first = base + (__ffs(mask) - 1);
    count += __popc(mask);
    if (count >= NSAMP) break;
  }
  lds_fence();
  int found = count < NSAMP ? count : NSAMP;
  int pad = (first >= 0) ? first : 0;
  int v = (lane < found) ? slots[wid][lane] : pad;
  bidx[sample * NSAMP + lane] = v;
}

// ---------------------------------------------------------------------------
// K4: grouped shared MLP via WMMA + neighbor maxpool.
//     8 waves/block, one sample point per wave. 72 v_wmma per sample.
//     A tile K order: [64 feats | dx dy dz | zeros..95] (W1p permuted to match)
// LDS: W1s 6144h | W2s 4096h | W3s 8192h | per wave: A(32x96)=3072h (aliased
//      by H2 32x64) + H(32x64)=2048h  -> 118784 B dynamic
// ---------------------------------------------------------------------------
__global__ void __launch_bounds__(256)
k_group_mlp(const float* __restrict__ xyz, const float* __restrict__ featsT,
            const float* __restrict__ new_xyz, const int* __restrict__ bidx,
            const _Float16* __restrict__ Wcat,  // W1p|W2p|W3p contiguous
            const float* __restrict__ b1, const float* __restrict__ b2,
            const float* __restrict__ b3, float* __restrict__ pooled) {
  extern __shared__ _Float16 smemh[];
  _Float16* W1s = smemh;            // [64][96]
  _Float16* W2s = W1s + 6144;       // [64][64]
  _Float16* W3s = W2s + 4096;       // [128][64]
  const int lane = threadIdx.x & 31;
  const int wid  = threadIdx.x >> 5;
  _Float16* A  = W3s + 8192 + wid * 5120; // 32 x 96 row-major
  _Float16* H  = A + 3072;                // 32 x 64 row-major
  _Float16* H2 = A;                       // reuse A after layer1

  // cooperative weight staging: 36864 B as b128 copies
  {
    const uint4* src = (const uint4*)Wcat;
    uint4* dst = (uint4*)smemh;
    for (int i = threadIdx.x; i < 2304; i += 256) dst[i] = src[i];
  }
  __syncthreads();

  const int sample = blockIdx.x * 8 + wid;
  const int b = sample / SQ, s = sample % SQ;

  // ---- gather: lane = neighbor j; A[j] = [f0..f63, dx,dy,dz, zeros]
  {
    int idxn = bidx[sample * NSAMP + lane];
    _Float16* Ar = A + lane * 96;
    const float4* fr = (const float4*)(featsT + (size_t)(b * NQ + idxn) * 64);
#pragma unroll
    for (int cc = 0; cc < 8; ++cc) {
      float4 f0 = fr[2 * cc], f1 = fr[2 * cc + 1];
      v8h h;
      h[0] = (_Float16)f0.x; h[1] = (_Float16)f0.y;
      h[2] = (_Float16)f0.z; h[3] = (_Float16)f0.w;
      h[4] = (_Float16)f1.x; h[5] = (_Float16)f1.y;
      h[6] = (_Float16)f1.z; h[7] = (_Float16)f1.w;
      *(v8h*)(Ar + cc * 8) = h;
    }
    float qx = new_xyz[(b * SQ + s) * 3 + 0];
    float qy = new_xyz[(b * SQ + s) * 3 + 1];
    float qz = new_xyz[(b * SQ + s) * 3 + 2];
    v8h hx;
    hx[0] = (_Float16)(xyz[(b * NQ + idxn) * 3 + 0] - qx);
    hx[1] = (_Float16)(xyz[(b * NQ + idxn) * 3 + 1] - qy);
    hx[2] = (_Float16)(xyz[(b * NQ + idxn) * 3 + 2] - qz);
#pragma unroll
    for (int c = 3; c < 8; ++c) hx[c] = (_Float16)0.f;
    *(v8h*)(Ar + 64) = hx;
    v8h hz;
#pragma unroll
    for (int c = 0; c < 8; ++c) hz[c] = (_Float16)0.f;
    *(v8h*)(Ar + 72) = hz;
    *(v8h*)(Ar + 80) = hz;
    *(v8h*)(Ar + 88) = hz;
  }
  lds_fence();

  // ---- layer 1: (32x96)x(96->64) -> H
  {
    v8f acc[2][4];
#pragma unroll
    for (int m = 0; m < 2; ++m)
#pragma unroll
      for (int n = 0; n < 4; ++n) acc[m][n] = zero8();
#pragma unroll
    for (int kc = 0; kc < 3; ++kc) {
      v16h a0 = load_a_frag(A, 0, kc * 32, 96);
      v16h a1 = load_a_frag(A, 16, kc * 32, 96);
#pragma unroll
      for (int n = 0; n < 4; ++n) {
        v16h bf = load_b_frag(W1s, kc * 32, n * 16, 96);
        acc[0][n] = __builtin_amdgcn_wmma_f32_16x16x32_f16(false, a0, false, bf,
                                                           (short)0, acc[0][n],
                                                           false, false);
        acc[1][n] = __builtin_amdgcn_wmma_f32_16x16x32_f16(false, a1, false, bf,
                                                           (short)0, acc[1][n],
                                                           false, false);
      }
    }
#pragma unroll
    for (int m = 0; m < 2; ++m)
#pragma unroll
      for (int n = 0; n < 4; ++n)
        store_c_relu_f16(H, acc[m][n], b1, m * 16, n * 16, 64);
  }
  lds_fence();

  // ---- layer 2: (32x64)x(64->64) -> H2
  {
    v8f acc[2][4];
#pragma unroll
    for (int m = 0; m < 2; ++m)
#pragma unroll
      for (int n = 0; n < 4; ++n) acc[m][n] = zero8();
#pragma unroll
    for (int kc = 0; kc < 2; ++kc) {
      v16h a0 = load_a_frag(H, 0, kc * 32, 64);
      v16h a1 = load_a_frag(H, 16, kc * 32, 64);
#pragma unroll
      for (int n = 0; n < 4; ++n) {
        v16h bf = load_b_frag(W2s, kc * 32, n * 16, 64);
        acc[0][n] = __builtin_amdgcn_wmma_f32_16x16x32_f16(false, a0, false, bf,
                                                           (short)0, acc[0][n],
                                                           false, false);
        acc[1][n] = __builtin_amdgcn_wmma_f32_16x16x32_f16(false, a1, false, bf,
                                                           (short)0, acc[1][n],
                                                           false, false);
      }
    }
#pragma unroll
    for (int m = 0; m < 2; ++m)
#pragma unroll
      for (int n = 0; n < 4; ++n)
        store_c_relu_f16(H2, acc[m][n], b2, m * 16, n * 16, 64);
  }
  lds_fence();

  // ---- layer 3: (32x64)x(64->128), fused bias+ReLU+maxpool over neighbors
#pragma unroll
  for (int nh = 0; nh < 2; ++nh) {
    v8f acc[2][4];
#pragma unroll
    for (int m = 0; m < 2; ++m)
#pragma unroll
      for (int n = 0; n < 4; ++n) acc[m][n] = zero8();
#pragma unroll
    for (int kc = 0; kc < 2; ++kc) {
      v16h a0 = load_a_frag(H2, 0, kc * 32, 64);
      v16h a1 = load_a_frag(H2, 16, kc * 32, 64);
#pragma unroll
      for (int n = 0; n < 4; ++n) {
        v16h bf = load_b_frag(W3s, kc * 32, nh * 64 + n * 16, 64);
        acc[0][n] = __builtin_amdgcn_wmma_f32_16x16x32_f16(false, a0, false, bf,
                                                           (short)0, acc[0][n],
                                                           false, false);
        acc[1][n] = __builtin_amdgcn_wmma_f32_16x16x32_f16(false, a1, false, bf,
                                                           (short)0, acc[1][n],
                                                           false, false);
      }
    }
#pragma unroll
    for (int n = 0; n < 4; ++n) {
      int col = (lane & 15) + 16 * n + 64 * nh;
      float bv = b3[col];
      float pm = -1e30f;
#pragma unroll
      for (int m = 0; m < 2; ++m)
#pragma unroll
        for (int v = 0; v < 8; ++v) {
          float x = acc[m][n][v] + bv;
          x = x > 0.f ? x : 0.f;
          pm = fmaxf(pm, x);
        }
      pm = fmaxf(pm, __shfl_xor(pm, 16, 32)); // combine the two row halves
      if (lane < 16) pooled[(size_t)sample * 128 + col] = pm;
    }
  }
}

// ---------------------------------------------------------------------------
// K5: aggregation GEMM (8192x128)x(128->128) via WMMA + confidence head.
//     8 waves/block, 16 rows per wave. Writes new_features (B,128,S) + scores.
// ---------------------------------------------------------------------------
__global__ void __launch_bounds__(256)
k_agg_conf(const float* __restrict__ pooled, const _Float16* __restrict__ WAp,
           const float* __restrict__ ba, const float* __restrict__ wc,
           const float* __restrict__ bc, float* __restrict__ nf_out,
           float* __restrict__ sc_out) {
  __shared__ _Float16 WAs[128 * 128];  // [n][k]
  {
    const uint4* src = (const uint4*)WAp;
    uint4* dst = (uint4*)WAs;
    for (int i = threadIdx.x; i < 2048; i += 256) dst[i] = src[i];
  }
  __syncthreads();

  const int lane = threadIdx.x & 31;
  const int wid  = threadIdx.x >> 5;
  const int hh   = lane >> 4;
  const int rowbase = blockIdx.x * 128 + wid * 16;

  v8f acc[8];
#pragma unroll
  for (int n = 0; n < 8; ++n) acc[n] = zero8();

#pragma unroll
  for (int kc = 0; kc < 4; ++kc) {
    // A fragment straight from global: two contiguous runs of 8 floats
    const int r = rowbase + (lane & 15);
    const float* p = pooled + (size_t)r * 128 + kc * 32 + hh * 8;
    float4 f0 = *(const float4*)(p + 0);
    float4 f1 = *(const float4*)(p + 4);
    float4 f2 = *(const float4*)(p + 16);
    float4 f3 = *(const float4*)(p + 20);
    v16h a;
    a[0] = (_Float16)f0.x;  a[1] = (_Float16)f0.y;
    a[2] = (_Float16)f0.z;  a[3] = (_Float16)f0.w;
    a[4] = (_Float16)f1.x;  a[5] = (_Float16)f1.y;
    a[6] = (_Float16)f1.z;  a[7] = (_Float16)f1.w;
    a[8] = (_Float16)f2.x;  a[9] = (_Float16)f2.y;
    a[10] = (_Float16)f2.z; a[11] = (_Float16)f2.w;
    a[12] = (_Float16)f3.x; a[13] = (_Float16)f3.y;
    a[14] = (_Float16)f3.z; a[15] = (_Float16)f3.w;
#pragma unroll
    for (int n = 0; n < 8; ++n) {
      v16h bf = load_b_frag(WAs, kc * 32, n * 16, 128);
      acc[n] = __builtin_amdgcn_wmma_f32_16x16x32_f16(false, a, false, bf,
                                                      (short)0, acc[n], false,
                                                      false);
    }
  }

  float sp[8];
#pragma unroll
  for (int v = 0; v < 8; ++v) sp[v] = 0.f;

#pragma unroll
  for (int n = 0; n < 8; ++n) {
    int col = (lane & 15) + 16 * n;
    float bv = ba[col];
    float wv = wc[col];
#pragma unroll
    for (int v = 0; v < 8; ++v) {
      int g = rowbase + v + 8 * hh;         // global row = b*S + s
      float x = acc[n][v] + bv;
      x = x > 0.f ? x : 0.f;
      int b = g >> 12, s = g & (SQ - 1);    // S = 4096
      nf_out[(size_t)b * 128 * SQ + (size_t)col * SQ + s] = x;
      sp[v] += x * wv;
    }
  }
#pragma unroll
  for (int v = 0; v < 8; ++v) {
#pragma unroll
    for (int m = 8; m >= 1; m >>= 1) sp[v] += __shfl_xor(sp[v], m, 32);
    if ((lane & 15) == 0) {
      int g = rowbase + v + 8 * hh;
      sc_out[g] = sp[v] + bc[0];
    }
  }
}

// ---------------------------------------------------------------------------
extern "C" void kernel_launch(void* const* d_in, const int* in_sizes, int n_in,
                              void* d_out, int out_size, void* d_ws,
                              size_t ws_size, hipStream_t stream) {
  const float* xyz   = (const float*)d_in[0];
  const float* feats = (const float*)d_in[1];
  const float* w1 = (const float*)d_in[2];
  const float* b1 = (const float*)d_in[3];
  const float* w2 = (const float*)d_in[4];
  const float* b2 = (const float*)d_in[5];
  const float* w3 = (const float*)d_in[6];
  const float* b3 = (const float*)d_in[7];
  const float* wa = (const float*)d_in[8];
  const float* ba = (const float*)d_in[9];
  const float* wc = (const float*)d_in[10];
  const float* bc = (const float*)d_in[11];

  char* ws = (char*)d_ws;
  float*     featsT = (float*)(ws + WS_FEATST);
  int*       sidx   = (int*)(ws + WS_SIDX);
  int*       bidx   = (int*)(ws + WS_BIDX);
  float*     pooled = (float*)(ws + WS_POOLED);
  _Float16*  W1p    = (_Float16*)(ws + WS_W1P);
  _Float16*  W2p    = (_Float16*)(ws + WS_W2P);
  _Float16*  W3p    = (_Float16*)(ws + WS_W3P);
  _Float16*  WAp    = (_Float16*)(ws + WS_WAP);

  float* out      = (float*)d_out;
  float* new_xyz  = out + OUT_NEWXYZ;
  float* new_feat = out + OUT_NEWFEAT;
  float* scores   = out + OUT_SCORES;

  const size_t grp_lds = 118784; // >64KB dynamic LDS (CDNA5 WGP has 320KB)
  (void)hipFuncSetAttribute((const void*)k_group_mlp,
                            hipFuncAttributeMaxDynamicSharedMemorySize,
                            (int)grp_lds);

  k_transpose_feats<<<(BQ * NQ * CIN) / 256, 256, 0, stream>>>(feats, featsT);
  k_pack_weights<<<64, 256, 0, stream>>>(w1, w2, w3, wa, W1p, W2p, W3p, WAp);
  k_fps<<<BQ, 1024, 0, stream>>>(xyz, sidx, new_xyz);
  k_ball_query<<<(BQ * SQ) / 8, 256, 0, stream>>>(xyz, new_xyz, bidx);
  k_group_mlp<<<(BQ * SQ) / 8, 256, grp_lds, stream>>>(
      xyz, featsT, new_xyz, bidx, W1p, b1, b2, b3, pooled);
  k_agg_conf<<<(BQ * SQ) / 128, 256, 0, stream>>>(pooled, WAp, ba, wc, bc,
                                                  new_feat, scores);
}
